// TorchMD_GN_14688788152851
// MI455X (gfx1250) — matrix operand order repaired and checked by reading, not compiled, not run on hardware
//
#include <hip/hip_runtime.h>
#include <hip/hip_bf16.h>
#include <math.h>

// ---------------------------------------------------------------------------
// TorchMD-GN fused pipeline for MI455X (gfx1250, wave32, WMMA 16x16x32 f16)
// Operands pre-swizzled into WMMA fragment layout: every lane loads its 16
// f16 elements as one contiguous 32B chunk (global_load_b128 / ds_load_b128).
// ---------------------------------------------------------------------------

typedef __attribute__((ext_vector_type(16))) _Float16 v16h;
typedef __attribute__((ext_vector_type(2)))  _Float16 v2h;
typedef __attribute__((ext_vector_type(8)))  float    v8f;
typedef __attribute__((ext_vector_type(4)))  float    v4f;

#define HIDDEN   128
#define NRBF     50
#define KRBF     64      // NUM_RBF padded to a multiple of 32 for WMMA K
#define NLAYERS  3
#define CUT_HI   5.0f

__device__ __forceinline__ v8f wmma16x16x32(v16h a, v16h b, v8f c) {
  return __builtin_amdgcn_wmma_f32_16x16x32_f16(false, a, false, b, (short)0, c,
                                                false, false);
}

__device__ __forceinline__ float silu(float v) {
  return v / (1.0f + __expf(-v));
}

// A-frag element i of lane (h*16+m) holds (row=m, k = h*8 + (i<8 ? i : i+8)).
// Inverse: given kk in [0,32): h=(kk>>3)&1, i=(kk&7)+((kk&16)?8:0)
__device__ __forceinline__ void a_swizzle(int kk, int& h, int& i) {
  h = (kk >> 3) & 1;
  i = (kk & 7) + ((kk & 16) ? 8 : 0);
}

// f32 A-fragment via two float4 pairs + convert (node GEMMs).
// p = A + row*lda + k0 + (lane>>4)*8
__device__ __forceinline__ v16h afrag_f32(const float* p) {
  v4f x0 = *(const v4f*)(p);
  v4f x1 = *(const v4f*)(p + 4);
  v4f x2 = *(const v4f*)(p + 16);
  v4f x3 = *(const v4f*)(p + 20);
  v16h a;
#pragma unroll
  for (int j = 0; j < 4; ++j) {
    a[j]      = (_Float16)x0[j];
    a[j + 4]  = (_Float16)x1[j];
    a[j + 8]  = (_Float16)x2[j];
    a[j + 12] = (_Float16)x3[j];
  }
  return a;
}

__device__ __forceinline__ v16h bfrag(const _Float16* __restrict__ WF,
                                      int kt, int nt, int lane) {
  return *(const v16h*)(WF + ((((size_t)kt * 8 + nt) * 32 + lane) << 4));
}

// Edge metadata for 8 consecutive edges, loaded as b128 vectors.
struct EdgeMeta { int s[8]; int d[8]; float c[8]; };
__device__ __forceinline__ EdgeMeta load_meta(const int* __restrict__ src,
                                              const int* __restrict__ dst,
                                              const float* __restrict__ C,
                                              int ebase) {
  EdgeMeta m;
  int4 s0 = *(const int4*)(src + ebase), s1 = *(const int4*)(src + ebase + 4);
  int4 d0 = *(const int4*)(dst + ebase), d1 = *(const int4*)(dst + ebase + 4);
  v4f  c0 = *(const v4f*)(C + ebase),   c1 = *(const v4f*)(C + ebase + 4);
  m.s[0]=s0.x; m.s[1]=s0.y; m.s[2]=s0.z; m.s[3]=s0.w;
  m.s[4]=s1.x; m.s[5]=s1.y; m.s[6]=s1.z; m.s[7]=s1.w;
  m.d[0]=d0.x; m.d[1]=d0.y; m.d[2]=d0.z; m.d[3]=d0.w;
  m.d[4]=d1.x; m.d[5]=d1.y; m.d[6]=d1.z; m.d[7]=d1.w;
#pragma unroll
  for (int j = 0; j < 4; ++j) { m.c[j] = c0[j]; m.c[j+4] = c1[j]; }
  return m;
}

// ---------------------------------------------------------------------------
// Kernel 1: edge_attr -> A-fragment layout; each thread computes a kk-pair
// (adjacent under the swizzle) and stores one b32.
//   eaF[tile][kstep][lane][i], tile = 16 edges, kstep in {0,1} (K=64).
// ---------------------------------------------------------------------------
__global__ void k_edge_attr(const float* __restrict__ ew,
                            const float* __restrict__ means,
                            const float* __restrict__ betas,
                            _Float16* __restrict__ eaF,
                            float* __restrict__ Cbuf, int E) {
  long long idx = (long long)blockIdx.x * blockDim.x + threadIdx.x;
  if (idx >= (long long)E * 32) return;
  int e = (int)(idx >> 5), kp = ((int)idx & 31) * 2;   // kp even, [0,64)
  float d = ew[e];
  float c = (d < CUT_HI) ? 0.5f * (__cosf(d * (3.14159265358979f / CUT_HI)) + 1.0f)
                         : 0.0f;
  if (kp == 0) Cbuf[e] = c;
  float ed = __expf(-d);                                // alpha=1, CUTOFF_LO=0
  v2h pr;
#pragma unroll
  for (int j = 0; j < 2; ++j) {
    int k = kp + j;
    float v = 0.0f;
    if (k < NRBF) {
      float t = ed - means[k];
      v = c * __expf(-betas[k] * t * t);               // includes cutoff
    }
    pr[j] = (_Float16)v;
  }
  int tile = e >> 4, m = e & 15;
  int kstep = kp >> 5, kk = kp & 31;
  int h, i; a_swizzle(kk, h, i);                        // kk even -> i even
  *(v2h*)(eaF + (size_t)tile * 1024 + kstep * 512 + (h * 16 + m) * 16 + i) = pr;
}

// ---------------------------------------------------------------------------
// Kernel 2: weight transpose  src[128][K] f32 -> B-fragment layout f16.
//   dst[((kt*8 + nt)*32 + lane)*16 + i]   (lane = h*16+n, k = kt*32+h*8+(i<8 ? i : i+8))
// ---------------------------------------------------------------------------
__global__ void k_wT(const float* __restrict__ src, _Float16* __restrict__ dst,
                     int K, int Kpad) {
  int idx = blockIdx.x * blockDim.x + threadIdx.x;
  if (idx >= Kpad * HIDDEN) return;
  int k = idx / HIDDEN, n = idx - k * HIDDEN;
  _Float16 v = (k < K) ? (_Float16)src[(size_t)n * K + k] : (_Float16)0.0f;
  int kt = k >> 5, koff = k & 31, h = (koff >> 3) & 1, i = (koff & 7) + ((koff & 16) ? 8 : 0);
  int nt = n >> 4, nn = n & 15;
  dst[(((size_t)kt * 8 + nt) * 32 + h * 16 + nn) * 16 + i] = v;
}

// ---------------------------------------------------------------------------
// Utility kernels (float4-vectorized streaming)
// ---------------------------------------------------------------------------
__global__ void k_gather_x(const float* __restrict__ emb,
                           const int* __restrict__ z,
                           float* __restrict__ x, int N) {
  long long idx = (long long)blockIdx.x * blockDim.x + threadIdx.x;
  if (idx >= (long long)N * 32) return;
  int a = (int)(idx >> 5), c4 = ((int)idx & 31) * 4;
  *(v4f*)(x + (size_t)a * HIDDEN + c4) =
      *(const v4f*)(emb + (size_t)z[a] * HIDDEN + c4);
}

__global__ void k_zero4(float* __restrict__ p, long long n4) {
  long long idx = (long long)blockIdx.x * blockDim.x + threadIdx.x;
  if (idx < n4) { v4f zz = {}; *(v4f*)(p + idx * 4) = zz; }
}

__global__ void k_concat(const float* __restrict__ x, const float* __restrict__ agg,
                         float* __restrict__ cat, int N) {
  long long idx = (long long)blockIdx.x * blockDim.x + threadIdx.x;
  if (idx >= (long long)N * 64) return;
  int a = (int)(idx >> 6), c4 = ((int)idx & 63) * 4;
  v4f v = (c4 < HIDDEN) ? *(const v4f*)(x + (size_t)a * HIDDEN + c4)
                        : *(const v4f*)(agg + (size_t)a * HIDDEN + (c4 - HIDDEN));
  *(v4f*)(cat + (size_t)a * 256 + c4) = v;
}

// ---------------------------------------------------------------------------
// Kernel 4: Neighbor-embedding edge pass, 32 edges (2 M-subtiles) per block.
//   W = (edge_attr @ projW + b) * C, masked (src!=dst); agg[src] += W * x[dst]
// ---------------------------------------------------------------------------
__global__ void __launch_bounds__(256)
k_ne_edge(const _Float16* __restrict__ eaF, const float* __restrict__ C,
          const int* __restrict__ src, const int* __restrict__ dst,
          const float* __restrict__ x,
          const _Float16* __restrict__ projWF,    // frag layout, K=64
          const float* __restrict__ projB,
          float* __restrict__ agg, int E) {
  const int tile0 = blockIdx.x * 2;
  const int lane  = threadIdx.x & 31;
  const int nt    = threadIdx.x >> 5;
  const int half_ = lane >> 4;

  v8f acc[2] = {{}, {}};
#pragma unroll
  for (int kt = 0; kt < 2; ++kt) {
    v16h b = bfrag(projWF, kt, nt, lane);
#pragma unroll
    for (int mt = 0; mt < 2; ++mt) {
      v16h a = *(const v16h*)(eaF + (size_t)(tile0 + mt) * 1024 + kt * 512 +
                              lane * 16);
      acc[mt] = wmma16x16x32(a, b, acc[mt]);
    }
  }

  const int n  = nt * 16 + (lane & 15);
  const float bv = projB[n];
#pragma unroll
  for (int mt = 0; mt < 2; ++mt) {
    const int ebase = (tile0 + mt) * 16 + 8 * half_;
    if (ebase + 8 <= E) {                        // fast path: vector metadata
      EdgeMeta em = load_meta(src, dst, C, ebase);
#pragma unroll
      for (int r = 0; r < 8; ++r) {
        if (em.s[r] != em.d[r]) {
          float w = (acc[mt][r] + bv) * em.c[r];
          atomicAdd(&agg[(size_t)em.s[r] * HIDDEN + n],
                    w * x[(size_t)em.d[r] * HIDDEN + n]);
        }
      }
    } else {                                     // ragged boundary
#pragma unroll
      for (int r = 0; r < 8; ++r) {
        int e = ebase + r;
        if (e < E) {
          int s = src[e], d = dst[e];
          if (s != d) {
            float w = (acc[mt][r] + bv) * C[e];
            atomicAdd(&agg[(size_t)s * HIDDEN + n],
                      w * x[(size_t)d * HIDDEN + n]);
          }
        }
      }
    }
  }
}

// ---------------------------------------------------------------------------
// Kernel 5: fused per-layer edge pass, 32 edges per block.
//   Wf = (silu(ea @ W0 + b0) @ W1 + b1) * C ; agg[src] += Wf * h[dst]
// GEMM1 output goes to LDS *in A-fragment layout* -> GEMM2 A = ds_load_b128.
// ---------------------------------------------------------------------------
__global__ void __launch_bounds__(256)
k_layer_edge(const _Float16* __restrict__ eaF, const float* __restrict__ C,
             const int* __restrict__ src, const int* __restrict__ dst,
             const float* __restrict__ h,
             const _Float16* __restrict__ W0F,   // frag layout, K=64
             const float* __restrict__ b0,
             const _Float16* __restrict__ W1F,   // frag layout, K=128
             const float* __restrict__ b1,
             float* __restrict__ agg, int E) {
  __shared__ _Float16 shA[2][4 * 32 * 16];       // 2 M-subtiles, K=128 frags

  const int tile0 = blockIdx.x * 2;
  const int lane  = threadIdx.x & 31;
  const int nt    = threadIdx.x >> 5;
  const int half_ = lane >> 4;
  const int n     = nt * 16 + (lane & 15);

  if ((tile0 + 2) * 16 < E)                      // uniform prefetch
    __builtin_prefetch(eaF + (size_t)(tile0 + 2) * 1024, 0, 0);

  // ---- GEMM1: ea[32x64] @ W0[64x128], SiLU -> LDS (A-frag layout) ---------
  v8f acc[2] = {{}, {}};
#pragma unroll
  for (int kt = 0; kt < 2; ++kt) {
    v16h b = bfrag(W0F, kt, nt, lane);
#pragma unroll
    for (int mt = 0; mt < 2; ++mt) {
      v16h a = *(const v16h*)(eaF + (size_t)(tile0 + mt) * 1024 + kt * 512 +
                              lane * 16);
      acc[mt] = wmma16x16x32(a, b, acc[mt]);
    }
  }
  {
    const float bv0 = b0[n];
    const int kt2 = n >> 5, kk2 = n & 31;        // col n becomes GEMM2 k-index
    int h2, i2; a_swizzle(kk2, h2, i2);
#pragma unroll
    for (int mt = 0; mt < 2; ++mt) {
#pragma unroll
      for (int r = 0; r < 8; ++r) {
        int m = r + 8 * half_;
        shA[mt][(kt2 * 32 + h2 * 16 + m) * 16 + i2] =
            (_Float16)silu(acc[mt][r] + bv0);
      }
    }
  }
  __syncthreads();

  // ---- GEMM2: T[32x128] (LDS frags) @ W1[128x128] -------------------------
  v8f acc2[2] = {{}, {}};
#pragma unroll
  for (int kt = 0; kt < 4; ++kt) {
    v16h b = bfrag(W1F, kt, nt, lane);
#pragma unroll
    for (int mt = 0; mt < 2; ++mt) {
      v16h a = *(const v16h*)&shA[mt][(kt * 32 + lane) * 16];
      acc2[mt] = wmma16x16x32(a, b, acc2[mt]);
    }
  }

  const float bv1 = b1[n];
#pragma unroll
  for (int mt = 0; mt < 2; ++mt) {
    const int ebase = (tile0 + mt) * 16 + 8 * half_;
    if (ebase + 8 <= E) {
      EdgeMeta em = load_meta(src, dst, C, ebase);
#pragma unroll
      for (int r = 0; r < 8; ++r) {
        float w = (acc2[mt][r] + bv1) * em.c[r];
        atomicAdd(&agg[(size_t)em.s[r] * HIDDEN + n],
                  w * h[(size_t)em.d[r] * HIDDEN + n]);
      }
    } else {
#pragma unroll
      for (int r = 0; r < 8; ++r) {
        int e = ebase + r;
        if (e < E) {
          float w = (acc2[mt][r] + bv1) * C[e];
          atomicAdd(&agg[(size_t)src[e] * HIDDEN + n],
                    w * h[(size_t)dst[e] * HIDDEN + n]);
        }
      }
    }
  }
}

// ---------------------------------------------------------------------------
// Kernel 6: node GEMM  out[M][128] = ep(A[M][K] @ W[K][128]); 32 rows/block.
// ---------------------------------------------------------------------------
__global__ void __launch_bounds__(256)
k_node_gemm(const float* __restrict__ A, int K,
            const _Float16* __restrict__ WF,     // frag layout
            const float* __restrict__ bias,
            const float* __restrict__ resid,
            float* __restrict__ out, int M, int act_silu) {
  const int row0  = blockIdx.x * 32;
  const int lane  = threadIdx.x & 31;
  const int nt    = threadIdx.x >> 5;
  const int half_ = lane >> 4;
  const int mloc  = lane & 15;

  int ra0 = row0 + mloc;       if (ra0 >= M) ra0 = M - 1;
  int ra1 = row0 + 16 + mloc;  if (ra1 >= M) ra1 = M - 1;

  v8f acc[2] = {{}, {}};
  for (int kt = 0; kt < (K >> 5); ++kt) {
    v16h b = bfrag(WF, kt, nt, lane);
    v16h a0 = afrag_f32(A + (size_t)ra0 * K + kt * 32 + half_ * 8);
    acc[0] = wmma16x16x32(a0, b, acc[0]);
    v16h a1 = afrag_f32(A + (size_t)ra1 * K + kt * 32 + half_ * 8);
    acc[1] = wmma16x16x32(a1, b, acc[1]);
  }

  const int n = nt * 16 + mloc;
  const float bv = bias ? bias[n] : 0.0f;
#pragma unroll
  for (int mt = 0; mt < 2; ++mt) {
#pragma unroll
    for (int r = 0; r < 8; ++r) {
      int m = row0 + mt * 16 + r + 8 * half_;
      if (m < M) {
        float v = acc[mt][r] + bv;
        if (act_silu) v = silu(v);
        if (resid)    v += resid[(size_t)m * HIDDEN + n];
        out[(size_t)m * HIDDEN + n] = v;
      }
    }
  }
}

// ---------------------------------------------------------------------------
// Host orchestration
// ---------------------------------------------------------------------------
extern "C" void kernel_launch(void* const* d_in, const int* in_sizes, int n_in,
                              void* d_out, int out_size, void* d_ws, size_t ws_size,
                              hipStream_t stream) {
  const int   N  = in_sizes[0];
  const int   E  = in_sizes[2];
  const int*  z          = (const int*)  d_in[0];
  const int*  ei         = (const int*)  d_in[1];
  const float* ew        = (const float*)d_in[2];
  const float* emb       = (const float*)d_in[3];
  const float* means     = (const float*)d_in[4];
  const float* betas     = (const float*)d_in[5];
  const float* ne_proj_w = (const float*)d_in[6];
  const float* ne_proj_b = (const float*)d_in[7];
  const float* ne_comb_w = (const float*)d_in[8];
  const float* ne_comb_b = (const float*)d_in[9];
  const float* mlp_w0    = (const float*)d_in[10];
  const float* mlp_b0    = (const float*)d_in[11];
  const float* mlp_w1    = (const float*)d_in[12];
  const float* mlp_b1    = (const float*)d_in[13];
  const float* lin1_w    = (const float*)d_in[14];
  const float* lin2_w    = (const float*)d_in[15];
  const float* lin2_b    = (const float*)d_in[16];
  const float* lin_w     = (const float*)d_in[17];
  const float* lin_b     = (const float*)d_in[18];

  const int* src = ei;
  const int* dst = ei + E;

  const int TB = 256;
  const int edgeBlocks = (E + 31) / 32;
  const int nodeBlocks = (N + 31) / 32;

  // --- workspace carve-up ---------------------------------------------------
  char*  base = (char*)d_ws;
  size_t off  = 0;
  auto carve = [&](size_t bytes) -> char* {
    char* p = base + off;
    off = (off + bytes + 255) & ~(size_t)255;
    return p;
  };
  _Float16* eaF   = (_Float16*)carve((size_t)edgeBlocks * 2 * 1024 * sizeof(_Float16));
  float*    Cb    = (float*)   carve((size_t)E * sizeof(float));
  float*    x     = (float*)   carve((size_t)N * HIDDEN * sizeof(float));
  float*    agg   = (float*)   carve((size_t)N * HIDDEN * sizeof(float));
  float*    tmp   = (float*)   carve((size_t)N * HIDDEN * sizeof(float));
  float*    hbuf  = (float*)   carve((size_t)N * HIDDEN * sizeof(float));
  float*    cat   = (float*)   carve((size_t)N * 256 * sizeof(float));
  _Float16* projW = (_Float16*)carve((size_t)KRBF * HIDDEN * sizeof(_Float16));
  _Float16* combW = (_Float16*)carve((size_t)256  * HIDDEN * sizeof(_Float16));
  _Float16* w0W   = (_Float16*)carve((size_t)NLAYERS * KRBF   * HIDDEN * sizeof(_Float16));
  _Float16* w1W   = (_Float16*)carve((size_t)NLAYERS * HIDDEN * HIDDEN * sizeof(_Float16));
  _Float16* l1W   = (_Float16*)carve((size_t)NLAYERS * HIDDEN * HIDDEN * sizeof(_Float16));
  _Float16* l2W   = (_Float16*)carve((size_t)NLAYERS * HIDDEN * HIDDEN * sizeof(_Float16));
  _Float16* lwW   = (_Float16*)carve((size_t)NLAYERS * HIDDEN * HIDDEN * sizeof(_Float16));
  (void)ws_size; (void)n_in; (void)out_size;

  // 1) edge features (fragment layout)
  {
    long long tot = (long long)E * 32;
    k_edge_attr<<<(unsigned)((tot + TB - 1) / TB), TB, 0, stream>>>(
        ew, means, betas, eaF, Cb, E);
  }
  // 2) weight transposes -> fragment-layout f16
  {
    int g50  = (KRBF * HIDDEN + TB - 1) / TB;
    int g128 = (HIDDEN * HIDDEN + TB - 1) / TB;
    int g256 = (256 * HIDDEN + TB - 1) / TB;
    k_wT<<<g50,  TB, 0, stream>>>(ne_proj_w, projW, NRBF, KRBF);
    k_wT<<<g256, TB, 0, stream>>>(ne_comb_w, combW, 256, 256);
    for (int l = 0; l < NLAYERS; ++l) {
      k_wT<<<g50,  TB, 0, stream>>>(mlp_w0 + (size_t)l*HIDDEN*NRBF,
                                    w0W + (size_t)l*KRBF*HIDDEN, NRBF, KRBF);
      k_wT<<<g128, TB, 0, stream>>>(mlp_w1 + (size_t)l*HIDDEN*HIDDEN,
                                    w1W + (size_t)l*HIDDEN*HIDDEN, HIDDEN, HIDDEN);
      k_wT<<<g128, TB, 0, stream>>>(lin1_w + (size_t)l*HIDDEN*HIDDEN,
                                    l1W + (size_t)l*HIDDEN*HIDDEN, HIDDEN, HIDDEN);
      k_wT<<<g128, TB, 0, stream>>>(lin2_w + (size_t)l*HIDDEN*HIDDEN,
                                    l2W + (size_t)l*HIDDEN*HIDDEN, HIDDEN, HIDDEN);
      k_wT<<<g128, TB, 0, stream>>>(lin_w + (size_t)l*HIDDEN*HIDDEN,
                                    lwW + (size_t)l*HIDDEN*HIDDEN, HIDDEN, HIDDEN);
    }
  }
  // 3) x = emb[z]
  {
    long long tot = (long long)N * 32;
    k_gather_x<<<(unsigned)((tot + TB - 1) / TB), TB, 0, stream>>>(emb, z, x, N);
  }
  const long long nh4 = (long long)N * 32;       // N*128/4 float4 stores
  // 4) Neighbor embedding
  {
    k_zero4<<<(unsigned)((nh4 + TB - 1) / TB), TB, 0, stream>>>(agg, nh4);
    k_ne_edge<<<edgeBlocks, TB, 0, stream>>>(eaF, Cb, src, dst, x, projW,
                                             ne_proj_b, agg, E);
    long long nc = (long long)N * 64;
    k_concat<<<(unsigned)((nc + TB - 1) / TB), TB, 0, stream>>>(x, agg, cat, N);
    k_node_gemm<<<nodeBlocks, TB, 0, stream>>>(cat, 256, combW, ne_comb_b,
                                               nullptr, x, N, 0);
  }
  // 5) interaction layers
  for (int l = 0; l < NLAYERS; ++l) {
    const _Float16* W0 = w0W + (size_t)l * KRBF * HIDDEN;
    const _Float16* W1 = w1W + (size_t)l * HIDDEN * HIDDEN;
    const _Float16* L1 = l1W + (size_t)l * HIDDEN * HIDDEN;
    const _Float16* L2 = l2W + (size_t)l * HIDDEN * HIDDEN;
    const _Float16* LW = lwW + (size_t)l * HIDDEN * HIDDEN;
    const float* b0 = mlp_b0 + (size_t)l * HIDDEN;
    const float* b1 = mlp_b1 + (size_t)l * HIDDEN;
    const float* b2 = lin2_b + (size_t)l * HIDDEN;
    const float* bl = lin_b  + (size_t)l * HIDDEN;

    k_node_gemm<<<nodeBlocks, TB, 0, stream>>>(x, HIDDEN, L1, nullptr, nullptr,
                                               hbuf, N, 0);
    k_zero4<<<(unsigned)((nh4 + TB - 1) / TB), TB, 0, stream>>>(agg, nh4);
    k_layer_edge<<<edgeBlocks, TB, 0, stream>>>(eaF, Cb, src, dst, hbuf,
                                                W0, b0, W1, b1, agg, E);
    k_node_gemm<<<nodeBlocks, TB, 0, stream>>>(agg, HIDDEN, L2, b2, nullptr,
                                               tmp, N, 1);
    k_node_gemm<<<nodeBlocks, TB, 0, stream>>>(tmp, HIDDEN, LW, bl, x, x, N, 0);
  }
  // 6) output
  hipMemcpyAsync(d_out, x, (size_t)N * HIDDEN * sizeof(float),
                 hipMemcpyDeviceToDevice, stream);
}
